// DepthGuidedFeatureVolume_21354577396153
// MI455X (gfx1250) — matrix operand order repaired and checked by reading, not compiled, not run on hardware
//
#include <hip/hip_runtime.h>

typedef __attribute__((ext_vector_type(16))) _Float16 v16h;
typedef __attribute__((ext_vector_type(8)))  float    v8f;

#define R   80
#define R3  512000
#define NV  4
#define C   32
#define H   64
#define W   64
#define HW  4096

struct __attribute__((aligned(16))) U128 { unsigned int x, y, z, w; };
union HV16 { v16h v; U128 q[2]; _Float16 h[16]; };

// Load a 16-bit A-matrix fragment from a 32-halfword LDS row.
// Lanes 0-15 (sel=0): K 0-15 in elems 0-15.
// Lanes 16-31 (sel=1): K 16-31 in elems 0-15.
__device__ __forceinline__ v16h load_a_frag(const _Float16* rowp, int sel, bool half_k) {
  HV16 u;
  u.q[0] = *(const U128*)(rowp + sel * 16);
  u.q[1] = *(const U128*)(rowp + sel * 16 + 8);
  if (half_k && sel) {
    U128 z0; z0.x = 0u; z0.y = 0u; z0.z = 0u; z0.w = 0u;
    u.q[0] = z0;
    u.q[1] = z0;
  }
  return u.v;
}

__global__ __launch_bounds__(256) void dgfv_wmma_kernel(
    const float* __restrict__ feats, const float* __restrict__ poses,
    const float* __restrict__ W1, const float* __restrict__ b1,
    const float* __restrict__ W2, const float* __restrict__ b2,
    const float* __restrict__ W3, const float* __restrict__ b3,
    float* __restrict__ out)
{
  // Per-wave private LDS staging (8 waves / block). No cross-wave sharing -> no barriers.
  __shared__ _Float16 s_tile[8][16][32];   // vf / h1 / h2 tiles (f16)
  __shared__ float    s_w[8][NV][16];      // per-view per-voxel weights
  __shared__ float    s_sw[8][16];         // sum of weights per voxel

  const int tid   = threadIdx.x;
  const int wid   = tid >> 5;
  const int lane  = tid & 31;
  const int n     = lane & 15;     // WMMA column (and A-row) id
  const int sel   = lane >> 4;     // half-select for K / row split
  const int ktop  = sel * 16;      // B-fragment K base
  const int rbase = sel * 8;       // D-tile row base

  // ---- B fragments + biases (branchless; amortized over 4 tiles x 4 views) ----
  // B layout: lane holds column n; elems 0..15 = K ktop..ktop+15.
  const bool w3live = (sel == 0) && (n < 8);
  v16h w1f0, w1f1, w2f, w3f;
#pragma unroll
  for (int j = 0; j < 16; ++j) {
    w1f0[j] = (_Float16)W1[(ktop + j) * 32 + n];
    w1f1[j] = (_Float16)W1[(ktop + j) * 32 + 16 + n];
    w2f[j]  = (_Float16)W2[(ktop + j) * 16 + n];
    const float w3v = W3[j * 8 + (n & 7)];        // always in-bounds, no divergence
    w3f[j]  = (_Float16)(w3live ? w3v : 0.0f);
  }
  const float c1a = b1[n];
  const float c1b = b1[16 + n];
  const float c2b = b2[n];
  const float b3v = b3[n & 7];                    // always in-bounds
  const float c3b = (n < 8) ? b3v : 0.0f;

#pragma unroll 1
  for (int t = 0; t < 4; ++t) {
    const int tile  = (blockIdx.x * 8 + wid) * 4 + t;
    const int mBase = tile * 16;   // voxel base, m = iz*R^2 + iy*R + ix (x fastest)

    // ---- view-invariant voxel coords for the sampling role (lane pair -> voxel) ----
    const int  rS = lane >> 1;          // voxel handled in sampling phase
    const int  chS = (lane & 1) * 16;   // channel half handled in sampling phase
    float pxS, pyS, pzS;
    {
      const int m  = mBase + rS;
      const int ix = m % R;
      const int iy = (m / R) % R;
      const int iz = m / (R * R);
      pxS = -1.0f + 2.0f * (float)ix / (float)(R - 1);
      pyS = -1.0f + 2.0f * (float)iy / (float)(R - 1);
      pzS = -1.0f + 2.0f * (float)iz / (float)(R - 1);
    }

    // ---- pre-pass: per-voxel view masks -> weights (all 32 lanes, dup writes ok) ----
    {
      const int m  = mBase + n;
      const int ix = m % R;
      const int iy = (m / R) % R;
      const int iz = m / (R * R);
      const float px = -1.0f + 2.0f * (float)ix / (float)(R - 1);
      const float py = -1.0f + 2.0f * (float)iy / (float)(R - 1);
      const float pz = -1.0f + 2.0f * (float)iz / (float)(R - 1);
      float mk[NV];
      float summ = 0.0f;
#pragma unroll
      for (int v = 0; v < NV; ++v) {
        const float* P = poses + v * 16;
        const float X = P[0] * px + P[1] * py + P[2]  * pz + P[3];
        const float Y = P[4] * px + P[5] * py + P[6]  * pz + P[7];
        const float Z = P[8] * px + P[9] * py + P[10] * pz + P[11];
        const float u  = X / Z;
        const float vy = Y / Z;
        const bool ok = (Z > 0.0f) && (u >= 0.0f) && (u <= 63.0f) &&
                        (vy >= 0.0f) && (vy <= 63.0f);
        mk[v] = ok ? 1.0f : 0.0f;
        summ += mk[v];
      }
      const float inv = 1.0f / (summ + 1e-8f);
#pragma unroll
      for (int v = 0; v < NV; ++v) s_w[wid][v][n] = mk[v] * inv;
      s_sw[wid][n] = summ * inv;
    }

    float S1[8], S2[8];
#pragma unroll
    for (int i = 0; i < 8; ++i) { S1[i] = 0.0f; S2[i] = 0.0f; }

#pragma unroll 1
    for (int v = 0; v < NV; ++v) {
      // ---- bilinear sampling: lane pair per voxel, 16 channels per lane ----
      {
        const float* P = poses + v * 16;
        const float X = P[0] * pxS + P[1] * pyS + P[2]  * pzS + P[3];
        const float Y = P[4] * pxS + P[5] * pyS + P[6]  * pzS + P[7];
        const float Z = P[8] * pxS + P[9] * pyS + P[10] * pzS + P[11];
        const bool zpos = (Z > 0.0f);
        const float u  = X / Z;
        const float vy = Y / Z;
        const float x0f = floorf(u);
        const float y0f = floorf(vy);
        const float wx1 = u - x0f,  wy1 = vy - y0f;
        const float wx0 = 1.0f - wx1, wy0 = 1.0f - wy1;
        const bool vx0 = (x0f >=  0.0f) && (x0f <= 63.0f);
        const bool vx1 = (x0f >= -1.0f) && (x0f <= 62.0f);
        const bool vY0 = (y0f >=  0.0f) && (y0f <= 63.0f);
        const bool vY1 = (y0f >= -1.0f) && (y0f <= 62.0f);
        const float w00 = (zpos && vx0 && vY0) ? wx0 * wy0 : 0.0f;
        const float w10 = (zpos && vx1 && vY0) ? wx1 * wy0 : 0.0f;
        const float w01 = (zpos && vx0 && vY1) ? wx0 * wy1 : 0.0f;
        const float w11 = (zpos && vx1 && vY1) ? wx1 * wy1 : 0.0f;
        const int x0c = (int)fminf(fmaxf(x0f,        0.0f), 63.0f);
        const int x1c = (int)fminf(fmaxf(x0f + 1.0f, 0.0f), 63.0f);
        const int y0c = (int)fminf(fmaxf(y0f,        0.0f), 63.0f);
        const int y1c = (int)fminf(fmaxf(y0f + 1.0f, 0.0f), 63.0f);
        const int f00 = y0c * 64 + x0c, f10 = y0c * 64 + x1c;
        const int f01 = y1c * 64 + x0c, f11 = y1c * 64 + x1c;

        HV16 row;
        const float* Fb = feats + (size_t)(v * C + chS) * HW;
#pragma unroll
        for (int c = 0; c < 16; ++c) {
          const float* Fc = Fb + c * HW;
          const float val = Fc[f00] * w00 + Fc[f10] * w10 +
                            Fc[f01] * w01 + Fc[f11] * w11;
          row.h[c] = (_Float16)val;
        }
        *(U128*)&s_tile[wid][rS][chS]     = row.q[0];
        *(U128*)&s_tile[wid][rS][chS + 8] = row.q[1];
      }

      // ---- layer 1: (16x32) @ W1(32x32), two 16-col tiles ----
      const v16h aA = load_a_frag(&s_tile[wid][n][0], sel, false);
      v8f cA, cB;
#pragma unroll
      for (int i = 0; i < 8; ++i) { cA[i] = c1a; cB[i] = c1b; }
      v8f d1a = __builtin_amdgcn_wmma_f32_16x16x32_f16(false, aA, false, w1f0, (short)0, cA, false, false);
      v8f d1b = __builtin_amdgcn_wmma_f32_16x16x32_f16(false, aA, false, w1f1, (short)0, cB, false, false);
#pragma unroll
      for (int i = 0; i < 8; ++i) {
        s_tile[wid][rbase + i][n]      = (_Float16)fmaxf(d1a[i], 0.0f);
        s_tile[wid][rbase + i][16 + n] = (_Float16)fmaxf(d1b[i], 0.0f);
      }

      // ---- layer 2: (16x32) @ W2(32x16) ----
      const v16h a2 = load_a_frag(&s_tile[wid][n][0], sel, false);
      v8f c2v;
#pragma unroll
      for (int i = 0; i < 8; ++i) c2v[i] = c2b;
      v8f d2 = __builtin_amdgcn_wmma_f32_16x16x32_f16(false, a2, false, w2f, (short)0, c2v, false, false);
#pragma unroll
      for (int i = 0; i < 8; ++i)
        s_tile[wid][rbase + i][n] = (_Float16)fmaxf(d2[i], 0.0f);

      // ---- layer 3: (16x16 zero-padded to K=32) @ W3(16x8 padded) ----
      const v16h a3 = load_a_frag(&s_tile[wid][n][0], sel, true);
      v8f c3v;
#pragma unroll
      for (int i = 0; i < 8; ++i) c3v[i] = c3b;
      v8f d3 = __builtin_amdgcn_wmma_f32_16x16x32_f16(false, a3, false, w3f, (short)0, c3v, false, false);

      // ---- accumulate weighted mean / second moment per (voxel,channel) ----
      const float* wp = &s_w[wid][v][rbase];
#pragma unroll
      for (int i = 0; i < 8; ++i) {
        const float wv = wp[i];
        const float h  = d3[i];
        S1[i] += wv * h;
        S2[i] += wv * h * h;
      }
    } // views

    // ---- finalize: mean = S1, var = S2 - mean^2 * (2 - sum_w) ----
    if (n < 8) {
      const float* swp = &s_sw[wid][rbase];
      float mv[8], vv[8];
#pragma unroll
      for (int i = 0; i < 8; ++i) {
        const float mean = S1[i];
        mv[i] = mean;
        vv[i] = S2[i] - mean * mean * (2.0f - swp[i]);
      }
      float* om = out + (size_t)n * R3 + mBase + rbase;
      float* ov = out + (size_t)(8 + n) * R3 + mBase + rbase;
      *(float4*)(om)     = make_float4(mv[0], mv[1], mv[2], mv[3]);
      *(float4*)(om + 4) = make_float4(mv[4], mv[5], mv[6], mv[7]);
      *(float4*)(ov)     = make_float4(vv[0], vv[1], vv[2], vv[3]);
      *(float4*)(ov + 4) = make_float4(vv[4], vv[5], vv[6], vv[7]);
    }
  } // tiles
}

extern "C" void kernel_launch(void* const* d_in, const int* in_sizes, int n_in,
                              void* d_out, int out_size, void* d_ws, size_t ws_size,
                              hipStream_t stream) {
  const float* feats = (const float*)d_in[0];
  const float* poses = (const float*)d_in[1];
  const float* W1    = (const float*)d_in[2];
  const float* b1    = (const float*)d_in[3];
  const float* W2    = (const float*)d_in[4];
  const float* b2    = (const float*)d_in[5];
  const float* W3    = (const float*)d_in[6];
  const float* b3    = (const float*)d_in[7];
  float* out = (float*)d_out;

  // 1000 blocks x 8 waves x 4 tiles x 16 voxels = 512000 voxels (exact cover)
  dim3 grid(1000), block(256);
  hipLaunchKernelGGL(dgfv_wmma_kernel, grid, block, 0, stream,
                     feats, poses, W1, b1, W2, b2, W3, b3, out);
}
